// SparseMoeBlock_67645734912502
// MI455X (gfx1250) — matrix operand order repaired and checked by reading, not compiled, not run on hardware
//
#include <hip/hip_runtime.h>
#include <hip/hip_bf16.h>
#include <cstdint>

#define HIDDEN 1024
#define FFN    2048
#define NEXP   8
#define TOPK   2
#define NTOK   2048   // BATCH*SEQ
#define TM     32     // tokens per expert tile

typedef __attribute__((ext_vector_type(16))) _Float16 v16h;
typedef __attribute__((ext_vector_type(8)))  float    v8f;

// ---------------------------------------------------------------------------
// Kernel 0: zero the combine-output region (it is accumulated with atomics)
// ---------------------------------------------------------------------------
__global__ void moe_zero_out(float* __restrict__ out) {
    int i = blockIdx.x * blockDim.x + threadIdx.x;
    if (i < NTOK * HIDDEN) out[i] = 0.0f;
}

// ---------------------------------------------------------------------------
// Kernel 1: router — logits, fp32 softmax, top-2, renormalize
// ---------------------------------------------------------------------------
__global__ void moe_router(const float* __restrict__ x,
                           const float* __restrict__ gw,
                           float* __restrict__ logits_out,
                           int*   __restrict__ top_idx,
                           float* __restrict__ top_w) {
    int t = blockIdx.x * blockDim.x + threadIdx.x;
    if (t >= NTOK) return;
    const float* xr = x + (size_t)t * HIDDEN;

    float acc[NEXP];
#pragma unroll
    for (int e = 0; e < NEXP; ++e) acc[e] = 0.0f;
    for (int d = 0; d < HIDDEN; ++d) {
        float xv = xr[d];
#pragma unroll
        for (int e = 0; e < NEXP; ++e)
            acc[e] = fmaf(xv, gw[e * HIDDEN + d], acc[e]);
    }

#pragma unroll
    for (int e = 0; e < NEXP; ++e) logits_out[t * NEXP + e] = acc[e];

    float mx = acc[0];
#pragma unroll
    for (int e = 1; e < NEXP; ++e) mx = fmaxf(mx, acc[e]);
    float p[NEXP]; float sum = 0.0f;
#pragma unroll
    for (int e = 0; e < NEXP; ++e) { p[e] = __expf(acc[e] - mx); sum += p[e]; }
    float inv = 1.0f / sum;
#pragma unroll
    for (int e = 0; e < NEXP; ++e) p[e] *= inv;

    float b0 = -1.0f, b1 = -1.0f; int i0 = 0, i1 = 0;
#pragma unroll
    for (int e = 0; e < NEXP; ++e) {
        if (p[e] > b0)      { b1 = b0; i1 = i0; b0 = p[e]; i0 = e; }
        else if (p[e] > b1) { b1 = p[e]; i1 = e; }
    }
    float wsum = b0 + b1;
    top_idx[t * TOPK + 0] = i0;
    top_idx[t * TOPK + 1] = i1;
    top_w[t * TOPK + 0] = b0 / wsum;
    top_w[t * TOPK + 1] = b1 / wsum;
}

// ---------------------------------------------------------------------------
// Kernel 2: deterministic per-expert token list build.
// One block of 256 threads = 8 wave32s; wave w compacts tokens for expert w
// via ballot + prefix popcount (stable token order, no atomics).
// ---------------------------------------------------------------------------
__global__ void moe_build_lists(const int*   __restrict__ top_idx,
                                const float* __restrict__ top_w,
                                int*   __restrict__ exp_tok,
                                float* __restrict__ exp_w,
                                int*   __restrict__ counts) {
    int e    = threadIdx.x >> 5;   // expert = wave id (wave32)
    int lane = threadIdx.x & 31;
    int offset = 0;
    for (int t0 = 0; t0 < NTOK; t0 += 32) {
        int t = t0 + lane;
        int i0 = top_idx[t * TOPK + 0];
        int i1 = top_idx[t * TOPK + 1];
        bool match = false; float w = 0.0f;
        if (i0 == e)      { match = true; w = top_w[t * TOPK + 0]; }
        else if (i1 == e) { match = true; w = top_w[t * TOPK + 1]; }
        unsigned m32 = (unsigned)__ballot(match);
        int pos = offset + __popc(m32 & ((1u << lane) - 1u));
        if (match) {
            exp_tok[e * NTOK + pos] = t;
            exp_w [e * NTOK + pos] = w;
        }
        offset += __popc(m32);
    }
    if (lane == 0) counts[e] = offset;
}

// ---------------------------------------------------------------------------
// Kernel 3: fused expert FFN. One block = one (expert, 32-token tile).
// Each of the 8 waves owns a unique 16-column group (stride 128) and computes
// BOTH 16-row halves of the token tile, so every streamed W1/W3/W2 fragment
// feeds 2 WMMAs (and the shared x/h A-fragments feed 4 in GEMM1).
// ---------------------------------------------------------------------------
__launch_bounds__(256)
__global__ void moe_experts(const float* __restrict__ x,
                            const float* __restrict__ w1,
                            const float* __restrict__ w3,
                            const float* __restrict__ w2,
                            const int*   __restrict__ exp_tok,
                            const float* __restrict__ exp_w,
                            const int*   __restrict__ counts,
                            float* __restrict__ out) {
    __shared__ _Float16 sx[TM * HIDDEN];   //  64 KB: x tile, f16
    __shared__ _Float16 sh[TM * FFN];      // 128 KB: swiglu(h) tile, f16
    __shared__ int      ltok[TM];
    __shared__ float    stw[TM];

    const int e    = blockIdx.y;
    const int tile = blockIdx.x;
    const int cnt  = counts[e];
    const int row0 = tile * TM;
    if (row0 >= cnt) return;                       // uniform per-block exit
    const int nvalid = min(TM, cnt - row0);

    const int tid = threadIdx.x;
    if (tid < TM) {
        int r = tid;
        ltok[r] = (r < nvalid) ? exp_tok[e * NTOK + row0 + r] : 0;
        stw [r] = (r < nvalid) ? exp_w [e * NTOK + row0 + r] : 0.0f;
    }
    __syncthreads();

    // Stage x tile: gather token rows, convert fp32 -> f16 (pad rows = 0)
    for (int i = tid; i < TM * HIDDEN; i += 256) {
        int r = i >> 10;            // / HIDDEN
        int c = i & (HIDDEN - 1);
        float xv = (r < nvalid) ? x[(size_t)ltok[r] * HIDDEN + c] : 0.0f;
        sx[i] = (_Float16)xv;
    }
    __syncthreads();

    const int lane  = tid & 31;
    const int wid   = tid >> 5;     // n-group 0..7
    const int laneM = lane & 15;
    const int khalf = lane >> 4;

    const float* w1e = w1 + (size_t)e * FFN * HIDDEN;
    const float* w3e = w3 + (size_t)e * FFN * HIDDEN;
    const float* w2e = w2 + (size_t)e * HIDDEN * FFN;

    const _Float16* arow0 = &sx[(laneM)      * HIDDEN];   // rows 0..15
    const _Float16* arow1 = &sx[(16 + laneM) * HIDDEN];   // rows 16..31

    // ---------------- GEMM1 + SwiGLU -> sh ----------------
    for (int nt = wid * 16; nt < FFN; nt += 128) {
        v8f accg0 = {}; v8f accg1 = {};
        v8f accu0 = {}; v8f accu1 = {};
        const float* w1r = w1e + (size_t)(nt + laneM) * HIDDEN;
        const float* w3r = w3e + (size_t)(nt + laneM) * HIDDEN;
#pragma unroll 2
        for (int k = 0; k < HIDDEN; k += 32) {
            __builtin_prefetch(w1r + k + 128, 0, 0);   // global_prefetch_b8
            __builtin_prefetch(w3r + k + 128, 0, 0);
            v16h a0, a1, bg, bu;
#pragma unroll
            for (int vv = 0; vv < 8; ++vv) {
                // A fragment (16x32 f16): lanes 0-15 M=0..15; K interleaved
                int kbA = ((vv >> 2) << 4) + (khalf << 3) + ((vv & 3) << 1);
                a0[2 * vv]     = arow0[k + kbA];
                a0[2 * vv + 1] = arow0[k + kbA + 1];
                a1[2 * vv]     = arow1[k + kbA];
                a1[2 * vv + 1] = arow1[k + kbA + 1];
                // B fragment (32x16 f16): N = laneM; K interleaved (same as A)
                int kbB = ((vv >> 2) << 4) + (khalf << 3) + ((vv & 3) << 1);
                bg[2 * vv]     = (_Float16)w1r[k + kbB];
                bg[2 * vv + 1] = (_Float16)w1r[k + kbB + 1];
                bu[2 * vv]     = (_Float16)w3r[k + kbB];
                bu[2 * vv + 1] = (_Float16)w3r[k + kbB + 1];
            }
            accg0 = __builtin_amdgcn_wmma_f32_16x16x32_f16(
                        false, a0, false, bg, (short)0, accg0, false, false);
            accg1 = __builtin_amdgcn_wmma_f32_16x16x32_f16(
                        false, a1, false, bg, (short)0, accg1, false, false);
            accu0 = __builtin_amdgcn_wmma_f32_16x16x32_f16(
                        false, a0, false, bu, (short)0, accu0, false, false);
            accu1 = __builtin_amdgcn_wmma_f32_16x16x32_f16(
                        false, a1, false, bu, (short)0, accu1, false, false);
        }
#pragma unroll
        for (int r = 0; r < 8; ++r) {
            // fast SiLU: g * rcp(1 + exp(-g)) — v_exp_f32 + v_rcp_f32
            float g0 = accg0[r], u0 = accu0[r];
            float g1 = accg1[r], u1 = accu1[r];
            float s0 = g0 * __builtin_amdgcn_rcpf(1.0f + __expf(-g0));
            float s1 = g1 * __builtin_amdgcn_rcpf(1.0f + __expf(-g1));
            int ml = r + (khalf << 3);               // C/D layout row
            sh[(ml)      * FFN + nt + laneM] = (_Float16)(s0 * u0);
            sh[(16 + ml) * FFN + nt + laneM] = (_Float16)(s1 * u1);
        }
    }
    __syncthreads();

    const _Float16* hrow0 = &sh[(laneM)      * FFN];
    const _Float16* hrow1 = &sh[(16 + laneM) * FFN];

    // ---------------- GEMM2 -> weighted atomic scatter ----------------
    for (int dt = wid * 16; dt < HIDDEN; dt += 128) {
        v8f acc0 = {}; v8f acc1 = {};
        const float* w2r = w2e + (size_t)(dt + laneM) * FFN;
#pragma unroll 2
        for (int k = 0; k < FFN; k += 32) {
            __builtin_prefetch(w2r + k + 128, 0, 0);
            v16h a0, a1, b;
#pragma unroll
            for (int vv = 0; vv < 8; ++vv) {
                int kbA = ((vv >> 2) << 4) + (khalf << 3) + ((vv & 3) << 1);
                a0[2 * vv]     = hrow0[k + kbA];
                a0[2 * vv + 1] = hrow0[k + kbA + 1];
                a1[2 * vv]     = hrow1[k + kbA];
                a1[2 * vv + 1] = hrow1[k + kbA + 1];
                int kbB = ((vv >> 2) << 4) + (khalf << 3) + ((vv & 3) << 1);
                b[2 * vv]     = (_Float16)w2r[k + kbB];
                b[2 * vv + 1] = (_Float16)w2r[k + kbB + 1];
            }
            acc0 = __builtin_amdgcn_wmma_f32_16x16x32_f16(
                       false, a0, false, b, (short)0, acc0, false, false);
            acc1 = __builtin_amdgcn_wmma_f32_16x16x32_f16(
                       false, a1, false, b, (short)0, acc1, false, false);
        }
#pragma unroll
        for (int r = 0; r < 8; ++r) {
            int ml0 = r + (khalf << 3);
            int ml1 = 16 + ml0;
            int col = dt + laneM;
            if (ml0 < nvalid)
                atomicAdd(&out[(size_t)ltok[ml0] * HIDDEN + col], acc0[r] * stw[ml0]);
            if (ml1 < nvalid)
                atomicAdd(&out[(size_t)ltok[ml1] * HIDDEN + col], acc1[r] * stw[ml1]);
        }
    }
}

// ---------------------------------------------------------------------------
// Host-side launcher
// ---------------------------------------------------------------------------
extern "C" void kernel_launch(void* const* d_in, const int* in_sizes, int n_in,
                              void* d_out, int out_size, void* d_ws, size_t ws_size,
                              hipStream_t stream) {
    (void)in_sizes; (void)n_in; (void)out_size; (void)ws_size;
    const float* x  = (const float*)d_in[0];   // [T, D]
    const float* gw = (const float*)d_in[1];   // [E, D]
    const float* w1 = (const float*)d_in[2];   // [E, F, D]
    const float* w2 = (const float*)d_in[3];   // [E, D, F]
    const float* w3 = (const float*)d_in[4];   // [E, F, D]

    float* out    = (float*)d_out;                    // [T*D]
    float* logits = out + (size_t)NTOK * HIDDEN;      // [T*E]

    // workspace layout (~160 KB)
    int*   top_idx = (int*)d_ws;                          // NTOK*TOPK
    float* top_w   = (float*)(top_idx + NTOK * TOPK);     // NTOK*TOPK
    int*   exp_tok = (int*)(top_w + NTOK * TOPK);         // NEXP*NTOK
    float* exp_w   = (float*)(exp_tok + NEXP * NTOK);     // NEXP*NTOK
    int*   counts  = (int*)(exp_w + NEXP * NTOK);         // NEXP

    moe_zero_out<<<(NTOK * HIDDEN + 255) / 256, 256, 0, stream>>>(out);
    moe_router<<<(NTOK + 255) / 256, 256, 0, stream>>>(x, gw, logits, top_idx, top_w);
    moe_build_lists<<<1, 256, 0, stream>>>(top_idx, top_w, exp_tok, exp_w, counts);

    dim3 grid(NTOK / TM, NEXP);   // worst-case tiles; blocks self-skip
    moe_experts<<<grid, 256, 0, stream>>>(x, w1, w3, w2, exp_tok, exp_w, counts, out);
}
